// ConceptCLIPLoss_58832462021083
// MI455X (gfx1250) — matrix operand order repaired and checked by reading, not compiled, not run on hardware
//
#include <hip/hip_runtime.h>
#include <hip/hip_bf16.h>
#include <math.h>

// ---------------------------------------------------------------------------
// ConceptCLIP loss for MI455X (gfx1250, wave32, WMMA).
// Dominant cost: A = einsum('mnd,vwd->mvnw') = 128x128 pair-GEMMs of
// (196x768)x(768x32)  ->  158 GFLOP, ~90MB inputs  => compute-bound.
// Route it through v_wmma_f32_16x16x32_f16 (f16 in, f32 accumulate).
// Staged f16 operands (38.6MB + 6.3MB) are L2-resident (192MB L2).
// ---------------------------------------------------------------------------

typedef _Float16     half16 __attribute__((ext_vector_type(16)));
typedef float        float8 __attribute__((ext_vector_type(8)));
typedef unsigned int uint4v __attribute__((ext_vector_type(4)));

#define B_  128
#define N_  196
#define D_  768
#define W_  32
#define KSTEPS (D_ / 32)   // 24 k-steps of 32
#define MTILES 13          // ceil(196/16)

__device__ __forceinline__ float log_sigmoid_f(float x) {
    // stable: log(sigmoid(x)) = min(x,0) - log1p(exp(-|x|))
    return fminf(x, 0.f) - log1pf(expf(-fabsf(x)));
}

__device__ __forceinline__ float temp_from(const float* p_ls) {
    float ls = fminf(fmaxf(p_ls[0], -10.f), 10.f);
    return expf(ls);
}

// -------- L2-normalize each 768-wide row, emit f16 ------------------------
__global__ __launch_bounds__(256)
void norm_rows_f16_kernel(const float* __restrict__ src, _Float16* __restrict__ dst)
{
    const int row = blockIdx.x;
    const float* s = src + (size_t)row * D_;
    float ss = 0.f;
    for (int i = threadIdx.x; i < D_; i += 256) { float x = s[i]; ss = fmaf(x, x, ss); }
    __shared__ float red[256];
    red[threadIdx.x] = ss;
    __syncthreads();
    for (int st = 128; st > 0; st >>= 1) {
        if (threadIdx.x < st) red[threadIdx.x] += red[threadIdx.x + st];
        __syncthreads();
    }
    const float inv = 1.f / fmaxf(sqrtf(red[0]), 1e-12f);
    _Float16* d = dst + (size_t)row * D_;
    for (int i = threadIdx.x; i < D_; i += 256) d[i] = (_Float16)(s[i] * inv);
}

// -------- SigLIP CLS logits: t * <img_m/|img_m|, txt_v/|txt_v|> + bias ----
__global__ __launch_bounds__(256)
void cls_logits_kernel(const float* __restrict__ img, const float* __restrict__ txt,
                       const float* __restrict__ p_ls, const float* __restrict__ p_lb,
                       float* __restrict__ out)
{
    const int idx = blockIdx.x * 256 + threadIdx.x;   // 0..16383
    const int m = idx >> 7, v = idx & 127;
    const float* a = img + (size_t)m * D_;
    const float* b = txt + (size_t)v * D_;
    float na = 0.f, nb = 0.f, d = 0.f;
    for (int i = 0; i < D_; ++i) {
        float x = a[i], y = b[i];
        na = fmaf(x, x, na); nb = fmaf(y, y, nb); d = fmaf(x, y, d);
    }
    const float dn = d / (fmaxf(sqrtf(na), 1e-12f) * fmaxf(sqrtf(nb), 1e-12f));
    out[idx] = temp_from(p_ls) * dn + p_lb[0];
}

// -------- RC-align: per (m,v) pair, 196x32 sims via WMMA, max over patches,
// masked mean over concepts, -> rc_logits[m][v] ----------------------------
__global__ __launch_bounds__(256)
void rc_wmma_kernel(const _Float16* __restrict__ patches,   // [128][196][768] f16, L2-normalized
                    const _Float16* __restrict__ concepts,  // [128][ 32][768] f16, L2-normalized
                    const int*      __restrict__ counts,    // [128]
                    const float*    __restrict__ p_ls,
                    const float*    __restrict__ p_lb,
                    float*          __restrict__ rc_logits) // [128][128]
{
    const int m    = blockIdx.x;
    const int v    = blockIdx.y;
    const int tid  = threadIdx.x;
    const int wave = tid >> 5;        // 0..7
    const int lane = tid & 31;
    const int ntile = wave & 1;       // which 16 concepts
    const int khalf = lane >> 4;      // lane-half selects K sub-ranges (ISA layout)
    const int l16   = lane & 15;

    __shared__ float smax[8][16];     // per-wave, per-(local n) max over this wave's M tiles

    // B fragment (32x16 f16): lane n = l16, per-lane K = {khalf*8..+7} U {16+khalf*8..+7}
    const _Float16* brow = concepts + (((size_t)v * W_) + (size_t)(ntile * 16 + l16)) * D_
                         + khalf * 8;

    float lanemax = -3.0e38f;
    const half16 zero16 = {};

    for (int mt = (wave >> 1); mt < MTILES; mt += 4) {
        const int  row   = mt * 16 + l16;
        const bool rowok = (row < N_);
        // A fragment (16x32 f16): lane row = l16, per-lane K = {khalf*8..+7} U {16+khalf*8..+7}
        const _Float16* arow = patches + (((size_t)m * N_) + (size_t)(rowok ? row : 0)) * D_
                             + khalf * 8;

        float8 acc = {0.f, 0.f, 0.f, 0.f, 0.f, 0.f, 0.f, 0.f};
        for (int ks = 0; ks < KSTEPS; ++ks) {
            const int kb = ks * 32;
            union { half16 h; uint4v u[2]; } A, Bm;
            A.h = zero16;
            if (rowok) {
                A.u[0] = *(const uint4v*)(arow + kb);        // K = kb + khalf*8 + 0..7
                A.u[1] = *(const uint4v*)(arow + kb + 16);   // K = kb + 16 + khalf*8 + 0..7
            }
            Bm.u[0] = *(const uint4v*)(brow + kb);           // K = kb + khalf*8 + 0..7
            Bm.u[1] = *(const uint4v*)(brow + kb + 16);      // K = kb + 16 + khalf*8 + 0..7
            acc = __builtin_amdgcn_wmma_f32_16x16x32_f16(false, A.h, false, Bm.h,
                                                         (short)0, acc, false, false);
        }
        // C layout: VGPR r holds M = mt*16 + r + 8*khalf for concept n = l16
#pragma unroll
        for (int r = 0; r < 8; ++r) {
            const int mg = mt * 16 + r + 8 * khalf;
            if (mg < N_) lanemax = fmaxf(lanemax, acc[r]);
        }
    }
    // merge the two M-halves (lanes l and l^16 hold the same concept n)
    lanemax = fmaxf(lanemax, __shfl_xor(lanemax, 16, 32));
    if (lane < 16) smax[wave][lane] = lanemax;
    __syncthreads();

    if (tid < 32) {
        const int n  = tid;          // global concept index 0..31
        const int nt = n >> 4;       // waves {nt, nt+2, nt+4, nt+6} own this ntile
        const int j  = n & 15;
        float mx = fmaxf(fmaxf(smax[nt][j],     smax[nt + 2][j]),
                         fmaxf(smax[nt + 4][j], smax[nt + 6][j]));
        const int cnt = counts[v];
        float val = (n < cnt) ? mx : 0.f;
#pragma unroll
        for (int off = 16; off > 0; off >>= 1) val += __shfl_down(val, off, 32);
        if (tid == 0) {
            const float S = val / (float)cnt;
            rc_logits[m * B_ + v] = temp_from(p_ls) * S + p_lb[0];
        }
    }
}

// -------- final reduction: sigmoid-contrastive losses + total --------------
__global__ __launch_bounds__(256)
void loss_kernel(const float* __restrict__ itl, const float* __restrict__ rcl,
                 float* __restrict__ out)
{
    float s1 = 0.f, s2 = 0.f;
    for (int i = threadIdx.x; i < B_ * B_; i += 256) {
        const int mm = i >> 7, vv = i & 127;
        const float z  = (mm == vv) ? 1.f : -1.f;
        const float x1 = z * fminf(fmaxf(itl[i], -50.f), 50.f);
        const float x2 = z * fminf(fmaxf(rcl[i], -50.f), 50.f);
        s1 -= log_sigmoid_f(x1);
        s2 -= log_sigmoid_f(x2);
    }
    __shared__ float r1[256], r2[256];
    r1[threadIdx.x] = s1; r2[threadIdx.x] = s2;
    __syncthreads();
    for (int st = 128; st > 0; st >>= 1) {
        if (threadIdx.x < st) {
            r1[threadIdx.x] += r1[threadIdx.x + st];
            r2[threadIdx.x] += r2[threadIdx.x + st];
        }
        __syncthreads();
    }
    if (threadIdx.x == 0) {
        const float it_loss = r1[0] / (float)(B_ * B_);
        const float rc_loss = r2[0] / (float)(B_ * B_);
        out[0] = it_loss + 0.5f * rc_loss;   // total (alpha = 0.5)
        out[1] = it_loss;
        out[2] = rc_loss;
    }
}

// ---------------------------------------------------------------------------
extern "C" void kernel_launch(void* const* d_in, const int* in_sizes, int n_in,
                              void* d_out, int out_size, void* d_ws, size_t ws_size,
                              hipStream_t stream)
{
    const float* img    = (const float*)d_in[0];   // (128, 768)
    const float* txt    = (const float*)d_in[1];   // (128, 768)
    const float* ptk    = (const float*)d_in[2];   // (128, 196, 768)
    const float* cpt    = (const float*)d_in[3];   // (128, 32, 768)
    const int*   counts = (const int*)  d_in[4];   // (128,)
    const float* p_ls   = (const float*)d_in[5];   // scalar
    const float* p_lb   = (const float*)d_in[6];   // scalar

    char*  ws  = (char*)d_ws;
    size_t off = 0;
    auto take = [&](size_t bytes) -> char* {
        char* p = ws + off;
        off = (off + bytes + 255) & ~(size_t)255;
        return p;
    };
    _Float16* patches_h  = (_Float16*)take((size_t)B_ * N_ * D_ * sizeof(_Float16)); // 38.6 MB
    _Float16* concepts_h = (_Float16*)take((size_t)B_ * W_ * D_ * sizeof(_Float16)); //  6.3 MB
    float*    it_logits  = (float*)   take((size_t)B_ * B_ * sizeof(float));
    float*    rc_logits  = (float*)   take((size_t)B_ * B_ * sizeof(float));

    // 1) normalize + downconvert operands of the big einsum
    norm_rows_f16_kernel<<<B_ * N_, 256, 0, stream>>>(ptk, patches_h);
    norm_rows_f16_kernel<<<B_ * W_, 256, 0, stream>>>(cpt, concepts_h);

    // 2) CLS sigmoid-contrastive logits (tiny, VALU)
    cls_logits_kernel<<<(B_ * B_) / 256, 256, 0, stream>>>(img, txt, p_ls, p_lb, it_logits);

    // 3) the 158-GFLOP region-concept alignment via WMMA
    rc_wmma_kernel<<<dim3(B_, B_), 256, 0, stream>>>(patches_h, concepts_h, counts,
                                                     p_ls, p_lb, rc_logits);

    // 4) losses
    loss_kernel<<<1, 256, 0, stream>>>(it_logits, rc_logits, (float*)d_out);
}
